// FeaturePropagationLayer_51599737094350
// MI455X (gfx1250) — compile-verified, matrix operand order checked by reading, not run on hardware
//
#include <hip/hip_runtime.h>
#include <hip/hip_bf16.h>

// ---- problem constants ----
#define NB   16
#define NC   1024
#define NF   4096
#define CIN  256
#define CSK  128
#define CH   256
#define ROWS (NB * NF)              // 65536 fine points
#define K1   (CIN + CSK)            // 384
#define HELEMS ((size_t)ROWS * CH)  // 16777216 output h elements

// padded LDS row strides (halves) to avoid bank conflicts:
// 392*2B = 784B = 196 dwords, 196 % 64 = 4 -> lane i starts at bank 4i (b128 = banks 4i..4i+3)
#define SROW1 392   // stride for 64x384 input tile
#define SROW2 264   // stride for 64x256 hidden tile (528B = 132 dwords, %64 = 4)

typedef _Float16 h16;
typedef __attribute__((ext_vector_type(16))) _Float16 v16h;
typedef __attribute__((ext_vector_type(8)))  float    v8f;

union V16 { v16h v; uint4 q[2]; };
union H8  { h16 h[8]; uint4 q; };

// ---------------------------------------------------------------------------
// 1) kNN (k=3) per cloud. Coarse points staged in LDS as (x,y,z,0.5*|p|^2).
//    Rank by monotonic key = 0.5|p|^2 - q.p  (d^2 = 2*key + |q|^2):
//    inner loop = 1 ds_load_b128 broadcast + 3 FMA + 1 cmp; insert path is
//    branch-skipped (rare after warmup).
// ---------------------------------------------------------------------------
__global__ void knn_kernel(const float* __restrict__ pos,
                           const float* __restrict__ pos_skip,
                           int* __restrict__ kidx,
                           float* __restrict__ kw) {
  __shared__ float4 pcs[NC];   // 16 KB
  int b = blockIdx.y;
  const float* pc = pos + (size_t)b * NC * 3;
  for (int j = threadIdx.x; j < NC; j += blockDim.x) {
    float px = pc[j * 3 + 0], py = pc[j * 3 + 1], pz = pc[j * 3 + 2];
    pcs[j] = make_float4(px, py, pz, 0.5f * (px * px + py * py + pz * pz));
  }
  __syncthreads();

  int f  = blockIdx.x * blockDim.x + threadIdx.x;
  int gf = b * NF + f;
  float qx = pos_skip[gf * 3 + 0];
  float qy = pos_skip[gf * 3 + 1];
  float qz = pos_skip[gf * 3 + 2];
  float nqx = -qx, nqy = -qy, nqz = -qz;
  float qq = qx * qx + qy * qy + qz * qz;

  float k0 = 3.4e38f, k1 = 3.4e38f, k2 = 3.4e38f;
  int   i0 = 0, i1 = 0, i2 = 0;
#pragma unroll 4
  for (int j = 0; j < NC; ++j) {
    float4 p = pcs[j];
    float key = fmaf(p.x, nqx, fmaf(p.y, nqy, fmaf(p.z, nqz, p.w)));
    if (key < k2) {            // rare: candidate enters top-3
      if (key < k0)      { k2 = k1; i2 = i1; k1 = k0; i1 = i0; k0 = key; i0 = j; }
      else if (key < k1) { k2 = k1; i2 = i1; k1 = key; i1 = j; }
      else               { k2 = key; i2 = j; }
    }
  }
  // recover squared distances (clamped; matches 1/max(d2,eps) weighting)
  float d0 = fmaxf(2.f * k0 + qq, 1e-16f);
  float d1 = fmaxf(2.f * k1 + qq, 1e-16f);
  float d2 = fmaxf(2.f * k2 + qq, 1e-16f);
  float w0 = 1.f / d0, w1 = 1.f / d1, w2 = 1.f / d2;
  float inv = 1.f / (w0 + w1 + w2);
  kidx[gf * 3 + 0] = b * NC + i0;
  kidx[gf * 3 + 1] = b * NC + i1;
  kidx[gf * 3 + 2] = b * NC + i2;
  kw[gf * 3 + 0] = w0 * inv;
  kw[gf * 3 + 1] = w1 * inv;
  kw[gf * 3 + 2] = w2 * inv;
}

// ---------------------------------------------------------------------------
// 2) Pack weights [K x 256] f32 row-major -> per-lane-contiguous f16 WMMA
//    B fragments. Fragment(kb, nt): 32 lanes x 16 halves contiguous.
// ---------------------------------------------------------------------------
__global__ void pack_w(const float* __restrict__ W, h16* __restrict__ Wp, int KB) {
  int tid = blockIdx.x * blockDim.x + threadIdx.x;
  int total = KB * 16 * 32 * 16;
  if (tid >= total) return;
  int h    = tid & 15;
  int lane = (tid >> 4) & 31;
  int frag = tid >> 9;
  int nt = frag & 15;
  int kb = frag >> 4;
  int kof;
  if (lane < 16) kof = (h < 8) ? h     : h + 8;
  else           kof = (h < 8) ? h + 8 : h + 16;
  int k = kb * 32 + kof;
  int n = nt * 16 + (lane & 15);
  Wp[tid] = (h16)W[(size_t)k * 256 + n];
}

// ---------------------------------------------------------------------------
// 3) Fused gather + GEMM1 + GEMM2 (one kernel, tiles resident in LDS).
//    Block = 256 threads = 8 waves, owns 64 consecutive fine rows.
//      stage A: interp(x)|x_skip -> LDS tile [64 x 384] f16 (padded rows)
//      stage B: GEMM1 (A from LDS, W1 frags from global/L2) -> acc regs
//      stage C: bias+ReLU -> LDS tile [64 x 256] f16
//      stage D: GEMM2 -> bias+ReLU -> f32 rows of d_out
//    Wave tiling: 4 M-strips x 2 N-halves, 16x128 per wave, 8 accumulator
//    tiles (64 VGPRs) -> no spills.
// ---------------------------------------------------------------------------
__global__ void __launch_bounds__(256)
fused_fp(const float* __restrict__ x,
         const float* __restrict__ xs,
         const int* __restrict__ kidx,
         const float* __restrict__ kw,
         const h16* __restrict__ W1p,
         const float* __restrict__ b1,
         const h16* __restrict__ W2p,
         const float* __restrict__ b2,
         float* __restrict__ out) {
  __shared__ h16 sbuf[64 * SROW1];   // 50176 B; hidden tile [64 x SROW2] aliases front

  int w      = threadIdx.x >> 5;     // wave 0..7
  int mstrip = w >> 1;               // 0..3
  int nhalf  = w & 1;                // 0..1
  int lane   = threadIdx.x & 31;
  int hi = lane >> 4;                // A layout: half h holds k 8h..8h+7 and 16+8h..16+8h+7
  int n0 = lane & 15;

  int row0g = blockIdx.x * 64;       // global row base of block

  // ---- stage A: build input tile in LDS (each wave fills 8 rows) ----
  {
    int c = lane * 8;
    for (int rr = 0; rr < 8; ++rr) {
      int local = w * 8 + rr;
      int r = row0g + local;
      int   i0 = kidx[r * 3 + 0], i1 = kidx[r * 3 + 1], i2 = kidx[r * 3 + 2];
      float w0 = kw[r * 3 + 0],  w1 = kw[r * 3 + 1],  w2 = kw[r * 3 + 2];
      const float4* p0 = (const float4*)(x + (size_t)i0 * CIN + c);
      const float4* p1 = (const float4*)(x + (size_t)i1 * CIN + c);
      const float4* p2 = (const float4*)(x + (size_t)i2 * CIN + c);
      float4 a0 = p0[0], a1 = p0[1];
      float4 b0 = p1[0], b1v = p1[1];
      float4 c0 = p2[0], c1 = p2[1];
      H8 o;
      o.h[0] = (h16)(w0 * a0.x + w1 * b0.x + w2 * c0.x);
      o.h[1] = (h16)(w0 * a0.y + w1 * b0.y + w2 * c0.y);
      o.h[2] = (h16)(w0 * a0.z + w1 * b0.z + w2 * c0.z);
      o.h[3] = (h16)(w0 * a0.w + w1 * b0.w + w2 * c0.w);
      o.h[4] = (h16)(w0 * a1.x + w1 * b1v.x + w2 * c1.x);
      o.h[5] = (h16)(w0 * a1.y + w1 * b1v.y + w2 * c1.y);
      o.h[6] = (h16)(w0 * a1.z + w1 * b1v.z + w2 * c1.z);
      o.h[7] = (h16)(w0 * a1.w + w1 * b1v.w + w2 * c1.w);
      *(uint4*)(sbuf + (size_t)local * SROW1 + c) = o.q;
      if (lane < 16) {  // skip features: 128 channels
        const float4* psk = (const float4*)(xs + (size_t)r * CSK + c);
        float4 s0 = psk[0], s1 = psk[1];
        H8 so;
        so.h[0] = (h16)s0.x; so.h[1] = (h16)s0.y; so.h[2] = (h16)s0.z; so.h[3] = (h16)s0.w;
        so.h[4] = (h16)s1.x; so.h[5] = (h16)s1.y; so.h[6] = (h16)s1.z; so.h[7] = (h16)s1.w;
        *(uint4*)(sbuf + (size_t)local * SROW1 + CIN + c) = so.q;
      }
    }
  }
  __syncthreads();

  v8f acc[8] = {};

  // ---- stage B: GEMM1, K = 384 (12 k-blocks) ----
  const h16* arow1 = sbuf + (size_t)(mstrip * 16 + n0) * SROW1;
  for (int kb = 0; kb < 12; ++kb) {
    V16 a;
    int koff = kb * 32 + hi * 8;
    a.q[0] = *(const uint4*)(arow1 + koff);
    a.q[1] = *(const uint4*)(arow1 + koff + 16);
    const uint4* bp = (const uint4*)W1p +
                      ((size_t)(kb * 16 + nhalf * 8) * 32 + lane) * 2;
#pragma unroll
    for (int nt = 0; nt < 8; ++nt) {
      V16 bfrag;
      bfrag.q[0] = bp[0];
      bfrag.q[1] = bp[1];
      bp += 64;
      acc[nt] = __builtin_amdgcn_wmma_f32_16x16x32_f16(
          false, a.v, false, bfrag.v, (short)0, acc[nt], false, false);
    }
  }
  __syncthreads();  // everyone done reading input tile before overwrite

  // ---- stage C: bias + ReLU -> hidden tile in LDS (f16) ----
  {
    int rloc = mstrip * 16 + hi * 8;
#pragma unroll
    for (int nt = 0; nt < 8; ++nt) {
      int col = (nhalf * 8 + nt) * 16 + n0;
      float bv = b1[col];
#pragma unroll
      for (int v = 0; v < 8; ++v) {
        float o = fmaxf(acc[nt][v] + bv, 0.f);
        sbuf[(size_t)(rloc + v) * SROW2 + col] = (h16)o;
      }
      acc[nt] = (v8f)(0.f);  // reset for GEMM2
    }
  }
  __syncthreads();

  // ---- stage D: GEMM2, K = 256 (8 k-blocks) ----
  const h16* arow2 = sbuf + (size_t)(mstrip * 16 + n0) * SROW2;
  for (int kb = 0; kb < 8; ++kb) {
    V16 a;
    int koff = kb * 32 + hi * 8;
    a.q[0] = *(const uint4*)(arow2 + koff);
    a.q[1] = *(const uint4*)(arow2 + koff + 16);
    const uint4* bp = (const uint4*)W2p +
                      ((size_t)(kb * 16 + nhalf * 8) * 32 + lane) * 2;
#pragma unroll
    for (int nt = 0; nt < 8; ++nt) {
      V16 bfrag;
      bfrag.q[0] = bp[0];
      bfrag.q[1] = bp[1];
      bp += 64;
      acc[nt] = __builtin_amdgcn_wmma_f32_16x16x32_f16(
          false, a.v, false, bfrag.v, (short)0, acc[nt], false, false);
    }
  }

  // ---- epilogue: bias + ReLU -> f32 output rows ----
  {
    int rbase = row0g + mstrip * 16 + hi * 8;
#pragma unroll
    for (int nt = 0; nt < 8; ++nt) {
      int col = (nhalf * 8 + nt) * 16 + n0;
      float bv = b2[col];
#pragma unroll
      for (int v = 0; v < 8; ++v) {
        float o = fmaxf(acc[nt][v] + bv, 0.f);
        out[(size_t)(rbase + v) * 256 + col] = o;
      }
    }
  }
}

// ---------------------------------------------------------------------------
// 4) Passthrough outputs: pos_skip (f32) and batch_skip (i32 bits).
// ---------------------------------------------------------------------------
__global__ void copy_tail(const float* __restrict__ ps,
                          const int* __restrict__ bs,
                          float* __restrict__ out) {
  int i = blockIdx.x * blockDim.x + threadIdx.x;
  const size_t base = HELEMS;
  if (i < ROWS * 3) {
    out[base + i] = ps[i];
  } else {
    int j = i - ROWS * 3;
    if (j < ROWS) ((int*)out)[base + (size_t)ROWS * 3 + j] = bs[j];
  }
}

// ---------------------------------------------------------------------------
// launch
// ---------------------------------------------------------------------------
extern "C" void kernel_launch(void* const* d_in, const int* in_sizes, int n_in,
                              void* d_out, int out_size, void* d_ws, size_t ws_size,
                              hipStream_t stream) {
  (void)in_sizes; (void)n_in; (void)out_size; (void)ws_size;
  const float* x   = (const float*)d_in[0];
  const float* pos = (const float*)d_in[1];
  const float* xs  = (const float*)d_in[2];
  const float* ps  = (const float*)d_in[3];
  const float* W1  = (const float*)d_in[4];
  const float* b1  = (const float*)d_in[5];
  const float* W2  = (const float*)d_in[6];
  const float* b2  = (const float*)d_in[7];
  const int*   bs  = (const int*)d_in[9];

  // workspace layout (bytes, 256-aligned): ~1.9 MB
  char* ws = (char*)d_ws;
  int*   kidx = (int*)(ws + 0);            //  786432 B
  float* kw   = (float*)(ws + 786432);     //  786432 B
  h16*   W1p  = (h16*)(ws + 1572864);      //  196608 B
  h16*   W2p  = (h16*)(ws + 1769472);      //  131072 B

  pack_w<<<384, 256, 0, stream>>>(W1, W1p, 12);            // 384x256
  pack_w<<<256, 256, 0, stream>>>(W2, W2p, 8);             // 256x256
  knn_kernel<<<dim3(NF / 256, NB), 256, 0, stream>>>(pos, ps, kidx, kw);
  fused_fp<<<ROWS / 64, 256, 0, stream>>>(x, xs, kidx, kw, W1p, b1, W2p, b2,
                                          (float*)d_out);
  copy_tail<<<(ROWS * 4) / 256, 256, 0, stream>>>(ps, bs, (float*)d_out);
}
